// GAT_41781441855680
// MI455X (gfx1250) — matrix operand order repaired and checked by reading, not compiled, not run on hardware
//
#include <hip/hip_runtime.h>
#include <hip/hip_bf16.h>

// ---------------------------------------------------------------------------
// GAT 2-layer forward for MI455X (gfx1250, wave32, WMMA).
// GEMMs use v_wmma_f32_16x16x32_bf16 with B matrices pre-packed into the
// per-lane fragment layout (contiguous k = 16*(lane/16) + t, same order as A) and staged in LDS.
// Edge softmax/scatter passes are HBM/atomic bound and use f32 atomics.
// ---------------------------------------------------------------------------

typedef float        v8f   __attribute__((ext_vector_type(8)));
typedef __bf16       v16bf __attribute__((ext_vector_type(16)));
typedef float        f32x4 __attribute__((ext_vector_type(4)));
typedef unsigned int u32x4 __attribute__((ext_vector_type(4)));

union BFrag { v16bf v; u32x4 q[2]; };

// Fixed problem shape (from reference): Fin = Fh = 512, H=8, C=64, Out=40.
#define FIN   512
#define FH    512
#define HEADS 8
#define HDIM  64
#define OUTC  40
#define OUTP  48          // OUTC padded to 3x16 for WMMA tiles
#define NEG_SLOPE 0.2f
#define BN_EPS    1e-5f

// ---------------- monotonic float <-> uint encoding for atomicMax ----------
__device__ __forceinline__ unsigned encodeF(float f) {
  unsigned u = __float_as_uint(f);
  return (u & 0x80000000u) ? ~u : (u | 0x80000000u);
}
__device__ __forceinline__ float decodeF(unsigned k) {
  unsigned u = (k & 0x80000000u) ? (k & 0x7fffffffu) : ~k;
  return __uint_as_float(u);
}

__device__ __forceinline__ void get_edge(const int* __restrict__ ei,
                                         long long E, long long e,
                                         int& s, int& d) {
  if (e < E) { s = ei[e]; d = ei[E + e]; }
  else       { s = d = (int)(e - E); }      // self loops appended
}

// ---------------- small utility kernels ------------------------------------
__global__ void k_fill_u32(unsigned* __restrict__ p, unsigned val, long long n) {
  long long i = (long long)blockIdx.x * blockDim.x + threadIdx.x;
  if (i < n) p[i] = val;
}

__global__ void k_init_out_bias(float* __restrict__ out, const float* __restrict__ b,
                                long long n, int C) {
  long long i = (long long)blockIdx.x * blockDim.x + threadIdx.x;
  if (i < n) out[i] = b[i % C];
}

// ---------------- weight packing into B-fragment layout --------------------
// Packed layout: p[((kt*NT16 + n16)*32 + lane)*16 + t], t=0..15 covering
// K = kt*32 + (lane<16 ? 0 : 16) + t, column n = n16*16 + (lane&15).
__global__ void k_pack_w1(const float* __restrict__ W, __bf16* __restrict__ P) {
  int id = blockIdx.x * blockDim.x + threadIdx.x;     // 16*32*32 = 16384
  if (id >= 16 * 32 * 32) return;
  int lane = id & 31;
  int n16  = (id >> 5) & 31;
  int kt   = id >> 10;
  int n    = n16 * 16 + (lane & 15);
  int kof  = kt * 32 + ((lane & 16) ? 16 : 0);
  __bf16* o = P + (size_t)id * 16;
  for (int t = 0; t < 16; ++t)
    o[t] = (__bf16)W[(size_t)(kof + t) * FH + n];
}

__global__ void k_pack_w2(const float* __restrict__ W, __bf16* __restrict__ P) {
  int id = blockIdx.x * blockDim.x + threadIdx.x;     // 16*3*32 = 1536
  if (id >= 16 * 3 * 32) return;
  int lane = id & 31;
  int n16  = (id >> 5) % 3;
  int kt   = id / 96;
  int n    = n16 * 16 + (lane & 15);
  int kof  = kt * 32 + ((lane & 16) ? 16 : 0);
  __bf16* o = P + (size_t)id * 16;
  for (int t = 0; t < 16; ++t)
    o[t] = (n < OUTC) ? (__bf16)W[(size_t)(kof + t) * OUTC + n] : (__bf16)0.0f;
}

// ---------------- GEMM1: h1[N,512] = x[N,512] @ W1, bf16 WMMA --------------
// Block = 256 threads = 8 waves; block tile 128(M) x 128(N); wave tile 16x128.
// Whole 512(K) x 128(N) B slice lives in LDS (128 KB).
extern __shared__ char smem_raw[];

__global__ void k_gemm1(const float* __restrict__ x, const __bf16* __restrict__ w1p,
                        float* __restrict__ h1, int Nrows) {
  __bf16* lds = (__bf16*)smem_raw;                    // 16*8*32*16 = 65536 bf16
  const int tid  = threadIdx.x;
  const int lane = tid & 31;
  const int wave = tid >> 5;
  const int mblk = blockIdx.x;
  const int nblk = blockIdx.y;                        // 0..3 (128 cols each)

  // Stage packed W slice: per kt, 8*32*16 = 4096 contiguous bf16 from global.
  for (int kt = 0; kt < 16; ++kt) {
    const u32x4* src = (const u32x4*)(w1p + ((size_t)(kt * 32 + nblk * 8)) * 512);
    u32x4* dst = (u32x4*)(lds + kt * 4096);
    for (int i = tid; i < 512; i += 256) dst[i] = src[i];
  }
  __syncthreads();

  int row = mblk * 128 + wave * 16 + (lane & 15);
  if (row >= Nrows) row = Nrows - 1;                  // clamp loads, guard stores
  const float* xrow = x + (size_t)row * FIN;

  v8f acc[8] = {};
  for (int kt = 0; kt < 16; ++kt) {
    int kb = kt * 32 + ((lane & 16) ? 16 : 0);
    f32x4 a0 = *(const f32x4*)(xrow + kb);
    f32x4 a1 = *(const f32x4*)(xrow + kb + 4);
    f32x4 a2 = *(const f32x4*)(xrow + kb + 8);
    f32x4 a3 = *(const f32x4*)(xrow + kb + 12);
    v16bf av;
    for (int i = 0; i < 4; ++i) {
      av[i]      = (__bf16)a0[i];
      av[4 + i]  = (__bf16)a1[i];
      av[8 + i]  = (__bf16)a2[i];
      av[12 + i] = (__bf16)a3[i];
    }
    const u32x4* bl = (const u32x4*)(lds + kt * 4096);
    for (int nf = 0; nf < 8; ++nf) {
      BFrag fb;
      fb.q[0] = bl[(nf * 32 + lane) * 2 + 0];
      fb.q[1] = bl[(nf * 32 + lane) * 2 + 1];
      acc[nf] = __builtin_amdgcn_wmma_f32_16x16x32_bf16(
          false, av, false, fb.v, (short)0, acc[nf], false, false);
    }
  }

  int rowbase = mblk * 128 + wave * 16 + ((lane & 16) ? 8 : 0);
  for (int nf = 0; nf < 8; ++nf) {
    int col = nblk * 128 + nf * 16 + (lane & 15);
    for (int r = 0; r < 8; ++r) {
      int rr = rowbase + r;
      if (rr < Nrows) h1[(size_t)rr * FH + col] = acc[nf][r];
    }
  }
}

// ---------------- GEMM2: h2[N,40] = hbn[N,512](bf16) @ W2 ------------------
__global__ void k_gemm2(const __bf16* __restrict__ hbn, const __bf16* __restrict__ w2p,
                        float* __restrict__ h2, int Nrows) {
  __bf16* lds = (__bf16*)smem_raw;                    // 16*3*32*16 = 24576 bf16
  const int tid  = threadIdx.x;
  const int lane = tid & 31;
  const int wave = tid >> 5;
  const int mblk = blockIdx.x;

  {
    const u32x4* src = (const u32x4*)w2p;
    u32x4* dst = (u32x4*)lds;
    for (int i = tid; i < 3072; i += 256) dst[i] = src[i];
  }
  __syncthreads();

  int row = mblk * 128 + wave * 16 + (lane & 15);
  if (row >= Nrows) row = Nrows - 1;
  const __bf16* arow = hbn + (size_t)row * FH;

  v8f acc[3] = {};
  for (int kt = 0; kt < 16; ++kt) {
    int kb = kt * 32 + ((lane & 16) ? 16 : 0);
    BFrag fa;
    fa.q[0] = *(const u32x4*)(arow + kb);             // K kb..kb+7
    fa.q[1] = *(const u32x4*)(arow + kb + 8);         // K kb+8..kb+15
    const u32x4* bl = (const u32x4*)lds;
    for (int nf = 0; nf < 3; ++nf) {
      BFrag fb;
      fb.q[0] = bl[((kt * 3 + nf) * 32 + lane) * 2 + 0];
      fb.q[1] = bl[((kt * 3 + nf) * 32 + lane) * 2 + 1];
      acc[nf] = __builtin_amdgcn_wmma_f32_16x16x32_bf16(
          false, fa.v, false, fb.v, (short)0, acc[nf], false, false);
    }
  }

  int rowbase = mblk * 128 + wave * 16 + ((lane & 16) ? 8 : 0);
  for (int nf = 0; nf < 3; ++nf) {
    int col = nf * 16 + (lane & 15);
    if (col >= OUTC) continue;
    for (int r = 0; r < 8; ++r) {
      int rr = rowbase + r;
      if (rr < Nrows) h2[(size_t)rr * OUTC + col] = acc[nf][r];
    }
  }
}

// ---------------- per-node attention logits --------------------------------
// One wave per node; lane covers 16 channels (head = lane/4); reduce x4 lanes.
__global__ void k_alpha_heads(const float* __restrict__ h, const float* __restrict__ asrc,
                              const float* __restrict__ adst, float* __restrict__ as,
                              float* __restrict__ ad, int Nrows) {
  int wv   = (blockIdx.x * blockDim.x + threadIdx.x) >> 5;
  int lane = threadIdx.x & 31;
  if (wv >= Nrows) return;
  const float* row = h + (size_t)wv * FH;
  int c0 = lane * 16;
  float ss = 0.f, sd = 0.f;
  for (int i = 0; i < 16; i += 4) {
    f32x4 hv = *(const f32x4*)(row + c0 + i);
    f32x4 av = *(const f32x4*)(asrc + c0 + i);
    f32x4 dv = *(const f32x4*)(adst + c0 + i);
    ss += hv[0]*av[0] + hv[1]*av[1] + hv[2]*av[2] + hv[3]*av[3];
    sd += hv[0]*dv[0] + hv[1]*dv[1] + hv[2]*dv[2] + hv[3]*dv[3];
  }
  ss += __shfl_xor(ss, 1, 32); ss += __shfl_xor(ss, 2, 32);
  sd += __shfl_xor(sd, 1, 32); sd += __shfl_xor(sd, 2, 32);
  if ((lane & 3) == 0) {
    as[(size_t)wv * HEADS + (lane >> 2)] = ss;
    ad[(size_t)wv * HEADS + (lane >> 2)] = sd;
  }
}

// One wave per node, single head, 40 channels.
__global__ void k_alpha_single(const float* __restrict__ h, const float* __restrict__ asrc,
                               const float* __restrict__ adst, float* __restrict__ as,
                               float* __restrict__ ad, int Nrows) {
  int wv   = (blockIdx.x * blockDim.x + threadIdx.x) >> 5;
  int lane = threadIdx.x & 31;
  if (wv >= Nrows) return;
  const float* row = h + (size_t)wv * OUTC;
  float v  = row[lane];
  float ss = v * asrc[lane];
  float sd = v * adst[lane];
  if (lane < 8) {
    float v2 = row[32 + lane];
    ss += v2 * asrc[32 + lane];
    sd += v2 * adst[32 + lane];
  }
  for (int o = 16; o > 0; o >>= 1) { ss += __shfl_xor(ss, o, 32); sd += __shfl_xor(sd, o, 32); }
  if (lane == 0) { as[wv] = ss; ad[wv] = sd; }
}

// ---------------- edge softmax passes --------------------------------------
__global__ void k_edge_max(const int* __restrict__ ei, long long E, long long Ep,
                           const float* __restrict__ as, const float* __restrict__ ad,
                           unsigned* __restrict__ m, int H) {
  long long gid = (long long)blockIdx.x * blockDim.x + threadIdx.x;
  if (gid >= Ep * H) return;
  long long e = gid / H; int h = (int)(gid % H);
  int s, d; get_edge(ei, E, e, s, d);
  float ev = as[(size_t)s * H + h] + ad[(size_t)d * H + h];
  ev = ev > 0.f ? ev : NEG_SLOPE * ev;
  atomicMax(&m[(size_t)d * H + h], encodeF(ev));
}

__global__ void k_edge_exp(const int* __restrict__ ei, long long E, long long Ep,
                           const float* __restrict__ as, const float* __restrict__ ad,
                           const unsigned* __restrict__ m, float* __restrict__ denom,
                           float* __restrict__ p, int H) {
  long long gid = (long long)blockIdx.x * blockDim.x + threadIdx.x;
  if (gid >= Ep * H) return;
  long long e = gid / H; int h = (int)(gid % H);
  int s, d; get_edge(ei, E, e, s, d);
  float ev = as[(size_t)s * H + h] + ad[(size_t)d * H + h];
  ev = ev > 0.f ? ev : NEG_SLOPE * ev;
  float pe = __expf(ev - decodeF(m[(size_t)d * H + h]));
  p[(size_t)e * H + h] = pe;
  atomicAdd(&denom[(size_t)d * H + h], pe);
}

// Layer-1 scatter: 128 threads/edge, each handles 4 channels (one head slice).
__global__ void k_scatter1(const int* __restrict__ ei, long long E, long long Ep,
                           const float* __restrict__ h1, const float* __restrict__ p,
                           const float* __restrict__ denom, float* __restrict__ out1) {
  long long gid = (long long)blockIdx.x * blockDim.x + threadIdx.x;
  if (gid >= Ep * 128) return;
  long long e = gid >> 7; int part = (int)(gid & 127);
  int s, d; get_edge(ei, E, e, s, d);
  int head = part >> 4;                               // (part*4)/64
  float w = p[(size_t)e * HEADS + head] / denom[(size_t)d * HEADS + head];
  f32x4 hv = *(const f32x4*)(h1 + (size_t)s * FH + part * 4);
  float* o = out1 + (size_t)d * FH + part * 4;
  atomicAdd(o + 0, hv[0] * w);
  atomicAdd(o + 1, hv[1] * w);
  atomicAdd(o + 2, hv[2] * w);
  atomicAdd(o + 3, hv[3] * w);
}

// Layer-2 scatter: one thread per (edge, channel).
__global__ void k_scatter2(const int* __restrict__ ei, long long E, long long Ep,
                           const float* __restrict__ h2, const float* __restrict__ p,
                           const float* __restrict__ denom, float* __restrict__ out) {
  long long gid = (long long)blockIdx.x * blockDim.x + threadIdx.x;
  if (gid >= Ep * OUTC) return;
  long long e = gid / OUTC; int c = (int)(gid % OUTC);
  int s, d; get_edge(ei, E, e, s, d);
  float w = p[e] / denom[d];
  atomicAdd(&out[(size_t)d * OUTC + c], h2[(size_t)s * OUTC + c] * w);
}

// ---------------- batchnorm ------------------------------------------------
__global__ void k_bn_stats(const float* __restrict__ out1, const float* __restrict__ b1,
                           float* __restrict__ sums, int Nrows) {
  int c = threadIdx.x + (blockIdx.y << 8);            // 512 cols
  long long r0 = (long long)blockIdx.x * 256;
  float bb = b1[c], s = 0.f, s2 = 0.f;
  for (int i = 0; i < 256; ++i) {
    long long r = r0 + i;
    if (r >= Nrows) break;
    float v = out1[r * FH + c] + bb;
    s += v; s2 += v * v;
  }
  atomicAdd(&sums[c], s);
  atomicAdd(&sums[FH + c], s2);
}

__global__ void k_bn_apply(const float* __restrict__ out1, const float* __restrict__ b1,
                           const float* __restrict__ sums, const float* __restrict__ gamma,
                           const float* __restrict__ beta, __bf16* __restrict__ hbn,
                           long long total, int Nrows) {
  long long gid = (long long)blockIdx.x * blockDim.x + threadIdx.x;
  if (gid >= total) return;
  int c = (int)(gid & (FH - 1));
  float inv_n = 1.0f / (float)Nrows;
  float mu  = sums[c] * inv_n;
  float var = sums[FH + c] * inv_n - mu * mu;
  float v = out1[gid] + b1[c];
  v = (v - mu) * rsqrtf(var + BN_EPS) * gamma[c] + beta[c];
  hbn[gid] = (__bf16)fmaxf(v, 0.f);
}

// ---------------------------------------------------------------------------
extern "C" void kernel_launch(void* const* d_in, const int* in_sizes, int n_in,
                              void* d_out, int out_size, void* d_ws, size_t ws_size,
                              hipStream_t stream) {
  const float* x      = (const float*)d_in[0];
  const int*   ei     = (const int*)  d_in[1];
  const float* W1     = (const float*)d_in[2];
  const float* a_src1 = (const float*)d_in[3];
  const float* a_dst1 = (const float*)d_in[4];
  const float* b1     = (const float*)d_in[5];
  const float* gamma  = (const float*)d_in[6];
  const float* beta   = (const float*)d_in[7];
  const float* W2     = (const float*)d_in[8];
  const float* a_src2 = (const float*)d_in[9];
  const float* a_dst2 = (const float*)d_in[10];
  const float* b2     = (const float*)d_in[11];
  float* out = (float*)d_out;

  const long long N  = in_sizes[0] / FIN;
  const long long E  = in_sizes[1] / 2;
  const long long Ep = E + N;

  // ---- workspace carve-up ----
  char* ws = (char*)d_ws;
  auto alloc = [&](size_t bytes) -> char* {
    char* p = ws; ws += (bytes + 255) & ~(size_t)255; return p;
  };
  float*    h1    = (float*)   alloc((size_t)N * FH * 4);
  float*    out1  = (float*)   alloc((size_t)N * FH * 4);
  __bf16*   hbn   = (__bf16*)  alloc((size_t)N * FH * 2);
  __bf16*   w1p   = (__bf16*)  alloc((size_t)FIN * FH * 2);
  __bf16*   w2p   = (__bf16*)  alloc((size_t)FIN * OUTP * 2);
  float*    as1   = (float*)   alloc((size_t)N * HEADS * 4);
  float*    ad1   = (float*)   alloc((size_t)N * HEADS * 4);
  unsigned* m1    = (unsigned*)alloc((size_t)N * HEADS * 4);
  float*    den1  = (float*)   alloc((size_t)N * HEADS * 4);
  float*    p1    = (float*)   alloc((size_t)Ep * HEADS * 4);
  float*    h2    = (float*)   alloc((size_t)N * OUTC * 4);
  float*    as2   = (float*)   alloc((size_t)N * 4);
  float*    ad2   = (float*)   alloc((size_t)N * 4);
  unsigned* m2    = (unsigned*)alloc((size_t)N * 4);
  float*    den2  = (float*)   alloc((size_t)N * 4);
  float*    p2    = (float*)   alloc((size_t)Ep * 4);
  float*    bnsum = (float*)   alloc((size_t)2 * FH * 4);

  auto blocks = [](long long work, int bs) { return (unsigned)((work + bs - 1) / bs); };

  // ---- weight packing (cheap, done every call for determinism) ----
  k_pack_w1<<<64, 256, 0, stream>>>(W1, w1p);
  k_pack_w2<<<6, 256, 0, stream>>>(W2, w2p);

  // ---- zero / init fills ----
  k_fill_u32<<<blocks((long long)N * FH, 256), 256, 0, stream>>>((unsigned*)out1, 0u, N * FH);
  k_fill_u32<<<blocks(N * HEADS, 256), 256, 0, stream>>>(m1, 0u, N * HEADS);
  k_fill_u32<<<blocks(N * HEADS, 256), 256, 0, stream>>>((unsigned*)den1, 0u, N * HEADS);
  k_fill_u32<<<blocks(N, 256), 256, 0, stream>>>(m2, 0u, N);
  k_fill_u32<<<blocks(N, 256), 256, 0, stream>>>((unsigned*)den2, 0u, N);
  k_fill_u32<<<blocks(2 * FH, 256), 256, 0, stream>>>((unsigned*)bnsum, 0u, 2 * FH);
  k_init_out_bias<<<blocks(N * OUTC, 256), 256, 0, stream>>>(out, b2, N * OUTC, OUTC);

  // ---- layer 1: GEMM + attention + scatter ----
  dim3 g1(blocks(N, 128), 4);
  k_gemm1<<<g1, 256, 128 * 1024, stream>>>(x, w1p, h1, (int)N);

  k_alpha_heads<<<blocks(N * 32, 256), 256, 0, stream>>>(h1, a_src1, a_dst1, as1, ad1, (int)N);
  k_edge_max<<<blocks(Ep * HEADS, 256), 256, 0, stream>>>(ei, E, Ep, as1, ad1, m1, HEADS);
  k_edge_exp<<<blocks(Ep * HEADS, 256), 256, 0, stream>>>(ei, E, Ep, as1, ad1, m1, den1, p1, HEADS);
  k_scatter1<<<blocks(Ep * 128, 256), 256, 0, stream>>>(ei, E, Ep, h1, p1, den1, out1);

  // ---- batchnorm + relu (writes bf16 activations for GEMM2) ----
  dim3 gbn(blocks(N, 256), 2);
  k_bn_stats<<<gbn, 256, 0, stream>>>(out1, b1, bnsum, (int)N);
  k_bn_apply<<<blocks(N * FH, 256), 256, 0, stream>>>(out1, b1, bnsum, gamma, beta, hbn,
                                                      N * FH, (int)N);

  // ---- layer 2: GEMM + attention + scatter ----
  k_gemm2<<<blocks(N, 128), 256, 48 * 1024, stream>>>(hbn, w2p, h2, (int)N);
  k_alpha_single<<<blocks(N * 32, 256), 256, 0, stream>>>(h2, a_src2, a_dst2, as2, ad2, (int)N);
  k_edge_max<<<blocks(Ep, 256), 256, 0, stream>>>(ei, E, Ep, as2, ad2, m2, 1);
  k_edge_exp<<<blocks(Ep, 256), 256, 0, stream>>>(ei, E, Ep, as2, ad2, m2, den2, p2, 1);
  k_scatter2<<<blocks(Ep * OUTC, 256), 256, 0, stream>>>(ei, E, Ep, h2, p2, den2, out);

  (void)n_in; (void)out_size; (void)ws_size;
}
